// RelativePartialMultiHeadSelfAttention_48687749268015
// MI455X (gfx1250) — matrix operand order repaired and checked by reading, not compiled, not run on hardware
//
#include <hip/hip_runtime.h>

typedef __attribute__((ext_vector_type(16))) __bf16 v16bf;
typedef __attribute__((ext_vector_type(8)))  float  v8f;

#define U_  1024
#define H_  16
#define DH_ 64
#define B_  2
#define Q_  1024
#define M_  1024
#define KL_ 2048
#define R_  2049

// ---------- helpers ----------
__device__ __forceinline__ __bf16 f2bf(float x) {
  unsigned r = (__float_as_uint(x) + 0x8000u) >> 16;   // round-to-nearest
  return __builtin_bit_cast(__bf16, (unsigned short)r);
}

// pack two floats into two bf16 in one dword (lo -> [15:0], hi -> [31:16])
__device__ __forceinline__ unsigned pack2bf(float lo, float hi) {
  unsigned a = (__float_as_uint(lo) + 0x8000u) >> 16;
  unsigned b = (__float_as_uint(hi) + 0x8000u) & 0xFFFF0000u;
  return a | b;
}

__device__ __forceinline__ float bperm_f(int srcLane, float x) {
  return __int_as_float(__builtin_amdgcn_ds_bpermute(srcLane << 2, __float_as_int(x)));
}

__device__ __forceinline__ v8f v8f_zero() {
  v8f z;
#pragma unroll
  for (int i = 0; i < 8; ++i) z[i] = 0.f;
  return z;
}

__device__ __forceinline__ v8f wmma_bf16(v16bf a, v16bf b, v8f c) {
  return __builtin_amdgcn_wmma_f32_16x16x32_bf16(false, a, false, b, (short)0, c,
                                                 false, false);
}

// A fragment: 16x32 (MxK), LDS row-major [m][k]
__device__ __forceinline__ v16bf frag_a(const __bf16* p, int stride, int lane) {
  int m = lane & 15, hi = lane >> 4;
  v16bf f;
#pragma unroll
  for (int v = 0; v < 8; ++v) {
    int k = ((v < 4) ? 0 : 16) + hi * 8 + 2 * (v & 3);
    f[2 * v]     = p[m * stride + k];
    f[2 * v + 1] = p[m * stride + k + 1];
  }
  return f;
}

// B fragment: 32x16 (KxN), LDS layout [n][k]
__device__ __forceinline__ v16bf frag_b(const __bf16* p, int stride, int lane) {
  int n = lane & 15, hi = lane >> 4;
  v16bf f;
#pragma unroll
  for (int v = 0; v < 8; ++v) {
    int k = ((v < 4) ? 0 : 16) + hi * 8 + 2 * (v & 3);
    f[2 * v]     = p[n * stride + k];
    f[2 * v + 1] = p[n * stride + k + 1];
  }
  return f;
}

// ---------- generic f32-in / bf16-WMMA / f32-out GEMM ----------
// C[z] = A[z] (Mrows x K, lda) * B (K x N, ldb, N given by gridDim.y*64)
__global__ __launch_bounds__(128)
void gemm_bf16_wmma(const float* __restrict__ A, long strideAz, int lda,
                    const float* __restrict__ Bm, int ldb,
                    float* __restrict__ C, long strideCz, int ldc,
                    int Mrows, int K)
{
  __shared__ __bf16 As[64][36];   // [m][k], 72B stride -> aligned dword pairs
  __shared__ __bf16 Bs[64][36];   // [n][k]
  const int tid  = threadIdx.x;
  const int lane = tid & 31;
  const int wv   = tid >> 5;
  const int bm   = blockIdx.x * 64;
  const int bn   = blockIdx.y * 64;
  const float* Ab = A + (long)blockIdx.z * strideAz;
  float*       Cb = C + (long)blockIdx.z * strideCz;

  v8f acc[4];
#pragma unroll
  for (int t = 0; t < 4; ++t) acc[t] = v8f_zero();

  const int  ar  = tid >> 1, ak = (tid & 1) * 16;
  const int  gr  = bm + ar;
  const float amsk = (gr < Mrows) ? 1.f : 0.f;          // branch-free M edge
  const long arow0 = (long)((gr < Mrows) ? gr : 0) * lda;
  const int  bk  = tid >> 2, bn0 = (tid & 3) * 16;

  for (int k0 = 0; k0 < K; k0 += 32) {
    {
      const float* arow = Ab + arow0 + k0 + ak;
#pragma unroll
      for (int i = 0; i < 8; ++i) {
        int d = 2 * i;
        *(unsigned*)&As[ar][ak + d] = pack2bf(arow[d] * amsk, arow[d + 1] * amsk);
      }
      const float* brow = Bm + (long)(k0 + bk) * ldb + bn;
#pragma unroll
      for (int i = 0; i < 16; ++i)
        Bs[bn0 + i][bk] = f2bf(brow[bn0 + i]);   // transpose while staging
    }
    __syncthreads();
    v16bf af = frag_a(&As[wv * 16][0], 36, lane);
#pragma unroll
    for (int t = 0; t < 4; ++t) {
      v16bf bfr = frag_b(&Bs[t * 16][0], 36, lane);
      acc[t] = wmma_bf16(af, bfr, acc[t]);
    }
    __syncthreads();
  }
  int n = lane & 15, hi = lane >> 4;
#pragma unroll
  for (int t = 0; t < 4; ++t)
#pragma unroll
    for (int v = 0; v < 8; ++v) {
      int m   = wv * 16 + v + 8 * hi;
      int grr = bm + m;
      if (grr < Mrows) Cb[(long)grr * ldc + bn + t * 16 + n] = acc[t][v];
    }
}

// ---------- fused flash attention (1 wave handles 16 q-rows of one (b,h)) ----------
__global__ __launch_bounds__(32)
void attn_kernel(const float* __restrict__ wq,
                 const float* __restrict__ wkv,
                 const float* __restrict__ wr,
                 const float* __restrict__ seg_mat,
                 const float* __restrict__ seg_emb,
                 const float* __restrict__ bias_c,
                 const float* __restrict__ bias_r,
                 const float* __restrict__ bias_s,
                 const float* __restrict__ perm,
                 float* __restrict__ outp)
{
  __shared__ __bf16 Qc[16][68];   // q + bias_context   [m][d]
  __shared__ __bf16 Qr[17][68];   // q + bias_relative  [m][d] (+1 row for wrap)
  __shared__ __bf16 Kl[16][68];   // key tile           [n][d]
  __shared__ __bf16 Ra[32][68];   // rel strip A        [col][d]
  __shared__ __bf16 Rb[32][68];   // rel strip B (wrap) [col][d]
  __shared__ __bf16 Vt[64][36];   // value tile T       [d][k]
  __shared__ __bf16 Pl[16][36];   // probabilities      [m][k]
  __shared__ float  qs[16][2];    // q_s . seg_embed

  const int lane = threadIdx.x;
  const int q0   = blockIdx.x * 16;
  const int h    = blockIdx.y;
  const int b    = blockIdx.z;
  const int n    = lane & 15, hi = lane >> 4;
  const int hoff = h * DH_;

  // stage Q tiles (bias added, bf16, packed dword stores)
  {
    int r = lane >> 1, d0 = (lane & 1) * 32;
    const float* qrow = wq + ((long)b * Q_ + q0 + r) * U_ + hoff + d0;
    const float* bc   = bias_c + hoff + d0;
    const float* br   = bias_r + hoff + d0;
#pragma unroll
    for (int i = 0; i < 16; ++i) {
      int d = 2 * i;
      float x0 = qrow[d], x1 = qrow[d + 1];
      *(unsigned*)&Qc[r][d0 + d] = pack2bf(x0 + bc[d], x1 + bc[d + 1]);
      *(unsigned*)&Qr[r][d0 + d] = pack2bf(x0 + br[d], x1 + br[d + 1]);
    }
    if (lane < 2) {                 // extra q-row for the wrap strip (mask, no branch)
      int d0b = lane * 32;
      int qe  = q0 + 16;
      float msk = (qe < Q_) ? 1.f : 0.f;
      const float* qrow2 = wq + ((long)b * Q_ + ((qe < Q_) ? qe : 0)) * U_ + hoff + d0b;
      const float* br2   = bias_r + hoff + d0b;
#pragma unroll
      for (int i = 0; i < 16; ++i) {
        int d = 2 * i;
        *(unsigned*)&Qr[16][d0b + d] =
            pack2bf((qrow2[d] + br2[d]) * msk, (qrow2[d + 1] + br2[d + 1]) * msk);
      }
    }
  }
  // qs_seg: lane -> (q = lane&15, s = lane>>4)
  {
    int m = lane & 15, s = lane >> 4;
    const float* qrow = wq + ((long)b * Q_ + q0 + m) * U_ + hoff;
    const float* se   = seg_emb + (long)s * U_ + hoff;
    float a = 0.f;
#pragma unroll
    for (int d = 0; d < DH_; ++d) a += (qrow[d] + bias_s[hoff + d]) * se[d];
    qs[m][s] = a;
  }
  __syncthreads();

  const v16bf aC0 = frag_a(&Qc[0][0], 68, lane);
  const v16bf aC1 = frag_a(&Qc[0][32], 68, lane);
  const v16bf aR0 = frag_a(&Qr[0][0], 68, lane);
  const v16bf aR1 = frag_a(&Qr[0][32], 68, lane);
  const v16bf aS0 = frag_a(&Qr[1][0], 68, lane);   // rows q+1
  const v16bf aS1 = frag_a(&Qr[1][32], 68, lane);

  v8f acc[4];
#pragma unroll
  for (int t = 0; t < 4; ++t) acc[t] = v8f_zero();
  float mrun[8], lrun[8];
#pragma unroll
  for (int v = 0; v < 8; ++v) { mrun[v] = -3.0e38f; lrun[v] = 0.f; }

  const float* wrb = wr + (long)b * R_ * U_;

  for (int j0 = 0; j0 < KL_; j0 += 32) {
    // prefetch next chunk's K/V rows into cache while we compute
    if (j0 + 32 < KL_) {
      const float* pk = wkv + ((long)b * KL_ + j0 + 32 + lane) * (2 * U_) + hoff;
      __builtin_prefetch(pk, 0, 1);
      __builtin_prefetch(pk + U_, 0, 1);
    }
    v8f sv[2];
#pragma unroll 1
    for (int half = 0; half < 2; ++half) {
      const int  jt   = j0 + 16 * half;
      const bool wrap = (jt - q0) >= 1010;          // some j-q >= 1025 in tile
      __syncthreads();
      {   // stage key tile  Kl[n=key][d]
        int r = lane >> 1, d0 = (lane & 1) * 32;
        const float* krow = wkv + ((long)b * KL_ + jt + r) * (2 * U_) + hoff + d0;
#pragma unroll
        for (int i = 0; i < 16; ++i) {
          int d = 2 * i;
          *(unsigned*)&Kl[r][d0 + d] = pack2bf(krow[d], krow[d + 1]);
        }
      }
      {   // stage rel strips: Y columns rbase..rbase+31 (branch-free clamp)
        int rbaseA = jt - q0 + 1009;                // (jt-q0) - 15 + 1024
        int rA = rbaseA + lane;
        float mA = (rA >= 0 && rA < R_) ? 1.f : 0.f;
        const float* rrow = wrb + (long)((rA >= 0 && rA < R_) ? rA : 0) * U_ + hoff;
#pragma unroll
        for (int i = 0; i < 32; ++i) {
          int d = 2 * i;
          *(unsigned*)&Ra[lane][d] = pack2bf(rrow[d] * mA, rrow[d + 1] * mA);
        }
        if (wrap) {
          int rB = jt - q0 - 1040 + lane;           // (jt-q0) - 15 - 1025 + lane
          float mB = (rB >= 0 && rB < R_) ? 1.f : 0.f;
          const float* rrow2 = wrb + (long)((rB >= 0 && rB < R_) ? rB : 0) * U_ + hoff;
#pragma unroll
          for (int i = 0; i < 32; ++i) {
            int d = 2 * i;
            *(unsigned*)&Rb[lane][d] = pack2bf(rrow2[d] * mB, rrow2[d + 1] * mB);
          }
        }
      }
      __syncthreads();
      // context scores
      v8f sc = v8f_zero();
      sc = wmma_bf16(aC0, frag_b(&Kl[0][0], 68, lane), sc);
      sc = wmma_bf16(aC1, frag_b(&Kl[0][32], 68, lane), sc);
      // relative score strips
      v8f yA[2], yB[2];
#pragma unroll
      for (int ct = 0; ct < 2; ++ct) {
        v8f y = v8f_zero();
        y = wmma_bf16(aR0, frag_b(&Ra[ct * 16][0], 68, lane), y);
        y = wmma_bf16(aR1, frag_b(&Ra[ct * 16][32], 68, lane), y);
        yA[ct] = y;
      }
      if (wrap) {
#pragma unroll
        for (int ct = 0; ct < 2; ++ct) {
          v8f y = v8f_zero();
          y = wmma_bf16(aS0, frag_b(&Rb[ct * 16][0], 68, lane), y);
          y = wmma_bf16(aS1, frag_b(&Rb[ct * 16][32], 68, lane), y);
          yB[ct] = y;
        }
      }
      // assemble scores: diagonal extraction + segment term
      v8f s;
#pragma unroll
      for (int v = 0; v < 8; ++v) {
        int m  = v + 8 * hi;
        int c  = n - m + 15;                 // strip column, 0..30
        int sl = (c & 15) + (hi << 4);
        float r0  = bperm_f(sl, yA[0][v]);
        float r1  = bperm_f(sl, yA[1][v]);
        float rel = (c < 16) ? r0 : r1;
        if (wrap) {
          float r2   = bperm_f(sl, yB[0][v]);
          float r3   = bperm_f(sl, yB[1][v]);
          float relB = (c < 16) ? r2 : r3;
          int dd = jt + n - q0 - m;          // j - q
          rel = (dd <= 1024) ? rel : relB;
        }
        const float* sm = seg_mat + (((long)b * Q_ + q0 + m) * KL_ + jt + n) * 2;
        float segv = sm[0] * qs[m][0] + sm[1] * qs[m][1];
        s[v] = (sc[v] + rel + segv) * 0.125f;   // 1/sqrt(64)
      }
      sv[half] = s;
    }
    // online softmax (rows live across 16-lane half-groups)
#pragma unroll
    for (int v = 0; v < 8; ++v) {
      float x = fmaxf(sv[0][v], sv[1][v]);
#pragma unroll
      for (int msk = 1; msk < 16; msk <<= 1) x = fmaxf(x, bperm_f(lane ^ msk, x));
      float mn      = fmaxf(mrun[v], x);
      float rescale = __expf(mrun[v] - mn);
      mrun[v] = mn;
      lrun[v] *= rescale;
#pragma unroll
      for (int t = 0; t < 4; ++t) acc[t][v] *= rescale;
      int m = v + 8 * hi;
      const float* prow = perm + ((long)b * Q_ + q0 + m) * KL_ + j0;
      float p0 = __expf(sv[0][v] - mn) * prow[n];
      float p1 = __expf(sv[1][v] - mn) * prow[16 + n];
      Pl[m][n]      = f2bf(p0);
      Pl[m][16 + n] = f2bf(p1);
      float rs = p0 + p1;
#pragma unroll
      for (int msk = 1; msk < 16; msk <<= 1) rs += bperm_f(lane ^ msk, rs);
      lrun[v] += rs;
    }
    __syncthreads();
    {   // stage V tile transposed: Vt[d][k]
      const float* vcol = wkv + ((long)b * KL_ + j0 + lane) * (2 * U_) + U_ + hoff;
#pragma unroll
      for (int d = 0; d < 64; ++d) Vt[d][lane] = f2bf(vcol[d]);
    }
    __syncthreads();
    v16bf pf = frag_a(&Pl[0][0], 36, lane);
#pragma unroll
    for (int t = 0; t < 4; ++t)
      acc[t] = wmma_bf16(pf, frag_b(&Vt[t * 16][0], 36, lane), acc[t]);
  }
  // normalize + write pre-projection output
#pragma unroll
  for (int t = 0; t < 4; ++t)
#pragma unroll
    for (int v = 0; v < 8; ++v) {
      int m = v + 8 * hi;
      float o = acc[t][v] / (lrun[v] + 1e-7f);
      outp[((long)b * Q_ + q0 + m) * U_ + hoff + t * 16 + n] = o;
    }
}

// ---------- host ----------
extern "C" void kernel_launch(void* const* d_in, const int* in_sizes, int n_in,
                              void* d_out, int out_size, void* d_ws, size_t ws_size,
                              hipStream_t stream) {
  (void)in_sizes; (void)n_in; (void)out_size; (void)ws_size;
  const float* inputs    = (const float*)d_in[0];
  const float* content   = (const float*)d_in[1];
  const float* memories  = (const float*)d_in[2];
  const float* relatives = (const float*)d_in[3];
  const float* seg_mat   = (const float*)d_in[4];
  const float* seg_emb   = (const float*)d_in[5];
  const float* bias_c    = (const float*)d_in[6];
  const float* bias_r    = (const float*)d_in[7];
  const float* bias_s    = (const float*)d_in[8];
  const float* perm      = (const float*)d_in[9];
  const float* kern      = (const float*)d_in[10];
  float* out = (float*)d_out;

  float* ws  = (float*)d_ws;
  float* wq  = ws;                                   // 2048*1024
  float* wkv = wq  + (long)B_ * Q_ * U_;             // 2*2048*2048
  float* wr  = wkv + (long)B_ * KL_ * 2 * U_;        // 4098*1024
  float* att = wr  + (long)B_ * R_ * U_;             // 2048*1024

  dim3 blk(128);
  // w_q = inputs @ k_q   (M=2048, N=1024)
  gemm_bf16_wmma<<<dim3(32, 16, 1), blk, 0, stream>>>(
      inputs, 0L, U_, kern, 5 * U_, wq, 0L, U_, B_ * Q_, U_);
  // memories @ k_kv -> w_kv rows [0,M)   (per batch, N=2048)
  gemm_bf16_wmma<<<dim3(16, 32, B_), blk, 0, stream>>>(
      memories, (long)M_ * U_, U_, kern + U_, 5 * U_,
      wkv, (long)KL_ * 2 * U_, 2 * U_, M_, U_);
  // content @ k_kv -> w_kv rows [M,KL)
  gemm_bf16_wmma<<<dim3(16, 32, B_), blk, 0, stream>>>(
      content, (long)Q_ * U_, U_, kern + U_, 5 * U_,
      wkv + (long)M_ * 2 * U_, (long)KL_ * 2 * U_, 2 * U_, Q_, U_);
  // relatives @ k_r -> w_r   (B*R = 4098 contiguous rows, ragged M)
  gemm_bf16_wmma<<<dim3((B_ * R_ + 63) / 64, 16, 1), blk, 0, stream>>>(
      relatives, 0L, U_, kern + 3 * U_, 5 * U_, wr, 0L, U_, B_ * R_, U_);
  // fused attention
  attn_kernel<<<dim3(Q_ / 16, H_, B_), dim3(32), 0, stream>>>(
      wq, wkv, wr, seg_mat, seg_emb, bias_c, bias_r, bias_s, perm, att);
  // final projection: att @ k_o -> out
  gemm_bf16_wmma<<<dim3(32, 16, 1), blk, 0, stream>>>(
      att, 0L, U_, kern + 4 * U_, 5 * U_, out, 0L, U_, B_ * Q_, U_);
}
